// CausalSelfAttention_29265907155084
// MI455X (gfx1250) — compile-verified, matrix operand order checked by reading, not run on hardware
//
#include <hip/hip_runtime.h>
#include <stdint.h>

#define T_SEQ 2048
#define C_DIM 2048
#define NHEAD 16
#define HSZ   128
#define G_KV  4
#define QKV_F 3072 /* (NH + 2G) * HS */

typedef __attribute__((ext_vector_type(16))) __bf16 bf16x16;
typedef __attribute__((ext_vector_type(8)))  __bf16 bf16x8;
typedef __attribute__((ext_vector_type(8)))  float  floatx8;
typedef __attribute__((ext_vector_type(4)))  unsigned u32x4;
typedef __attribute__((ext_vector_type(8)))  int      i32x8;
typedef __attribute__((ext_vector_type(4)))  int      i32x4;

union Frag { bf16x16 v; bf16x8 h[2]; };

__device__ __forceinline__ __bf16 f2bf(float f) {
  union { float f; unsigned u; } c; c.f = f;
  unsigned r = c.u + 0x7FFFu + ((c.u >> 16) & 1u);
  unsigned short hs = (unsigned short)(r >> 16);
  return __builtin_bit_cast(__bf16, hs);
}

__device__ __forceinline__ float redmax16(float v) {
  v = fmaxf(v, __shfl_xor(v, 1, 32));
  v = fmaxf(v, __shfl_xor(v, 2, 32));
  v = fmaxf(v, __shfl_xor(v, 4, 32));
  v = fmaxf(v, __shfl_xor(v, 8, 32));
  return v;
}
__device__ __forceinline__ float redsum16(float v) {
  v += __shfl_xor(v, 1, 32);
  v += __shfl_xor(v, 2, 32);
  v += __shfl_xor(v, 4, 32);
  v += __shfl_xor(v, 8, 32);
  return v;
}

// ---- Tensor Data Mover: 2D tile (tile_cols x tile_rows) global -> LDS ----
// D# per CDNA5 ISA. data_size=2B. LDS rows padded by 16B
// every (1<<pad_interval) QWORDs. Tracked on TENSORcnt. EXEC-independent;
// call from a single wave. 6-arg builtin (clang-23 / therock-10.0 headers).
__device__ __forceinline__ void tdm_load_2d(unsigned lds_addr, const void* gptr,
                                            int tile_cols, int tile_rows,
                                            int row_stride_elems, int pad_interval) {
  unsigned long long ga = (unsigned long long)(uintptr_t)gptr;
  u32x4 g0;
  g0[0] = 1u;                                   // count=1, user mode
  g0[1] = lds_addr;                             // LDS byte address
  g0[2] = (unsigned)ga;                         // global_addr[31:0]
  g0[3] = (unsigned)((ga >> 32) & 0x01FFFFFFull) | (2u << 30); // addr[56:32], type=2
  i32x8 g1;
  g1[0] = (1 << 16)               /* data_size = 2 bytes */
        | (1 << 20)               /* pad_enable */
        | (pad_interval << 22)    /* pad every 2^pi QWORDs */
        | (3 << 25);              /* pad_amount = 4 DWORDs (16B) */
  g1[1] = 0;                      // atomic_barrier=0, tensor_dim0[15:0]=0
  g1[2] = 16;                     // tensor_dim0 = 1<<20 (hi16=16); tensor_dim1 lo16=0
  g1[3] = 16 | (tile_cols << 16); // tensor_dim1 = 1<<20; tile_dim0
  g1[4] = tile_rows & 0xFFFF;     // tile_dim1; tile_dim2=0 (2D)
  g1[5] = row_stride_elems;       // tensor_dim0_stride[31:0]
  g1[6] = 0;
  g1[7] = 0;
  i32x4 z4; z4[0] = 0; z4[1] = 0; z4[2] = 0; z4[3] = 0;
  i32x8 z8; z8[0] = 0; z8[1] = 0; z8[2] = 0; z8[3] = 0;
  z8[4] = 0; z8[5] = 0; z8[6] = 0; z8[7] = 0;
  __builtin_amdgcn_tensor_load_to_lds(g0, g1, z4, z4, z8, 0);
}

// ---------------- fp32 -> bf16 conversion ----------------
__global__ __launch_bounds__(256)
void f32_to_bf16_kernel(const float* __restrict__ in, __bf16* __restrict__ out, int n) {
  int i = blockIdx.x * 256 + threadIdx.x;
  int stride = gridDim.x * 256;
  for (; i < n; i += stride) out[i] = f2bf(in[i]);
}

// ---------------- bf16 GEMM: C[M,N] = A[M,K] * Bw[N,K]^T ----------------
// 256 threads = 8 waves; block tile 128x128, BK=32; wave tile 64x32 (4x2 WMMAs).
// TDM double-buffered staging: wave 0 posts next tiles, all waves compute.
#define LDSTRIDE 40  // bf16 elements per LDS row (80B, 16B aligned, conflict-free)
#define TILEB (128 * LDSTRIDE) // elements per staging buffer
__global__ __launch_bounds__(256)
void gemm_bf16_f32(const __bf16* __restrict__ A, const __bf16* __restrict__ Bw,
                   float* __restrict__ C, int M, int N, int K) {
  __shared__ __align__(16) __bf16 As[2 * TILEB];
  __shared__ __align__(16) __bf16 Bs[2 * TILEB];

  int tid  = threadIdx.x;
  int wave = tid >> 5, lane = tid & 31;
  int n16  = lane & 15, half = lane >> 4;
  int m0 = blockIdx.y * 128, n0 = blockIdx.x * 128;
  int wy = wave & 1;   // 2 groups of 64 rows
  int wx = wave >> 1;  // 4 groups of 32 cols

  floatx8 acc[4][2];
  for (int i = 0; i < 4; ++i) for (int j = 0; j < 2; ++j) acc[i][j] = (floatx8)0.f;

  const __bf16* Abase = A  + (size_t)m0 * K;
  const __bf16* Bbase = Bw + (size_t)n0 * K;
  unsigned ldsA = (unsigned)(uintptr_t)&As[0];
  unsigned ldsB = (unsigned)(uintptr_t)&Bs[0];
  int ktiles = K >> 5;

  if (wave == 0) { // prime buffer 0
    tdm_load_2d(ldsA, Abase, 32, 128, K, 3);
    tdm_load_2d(ldsB, Bbase, 32, 128, K, 3);
  }

  for (int kt = 0; kt < ktiles; ++kt) {
    int cur = kt & 1;
    if (wave == 0) __builtin_amdgcn_s_wait_tensorcnt(0);
    __syncthreads(); // buffer `cur` ready for all waves; prev compute done
    if (wave == 0 && kt + 1 < ktiles) { // post next tiles into other buffer
      tdm_load_2d(ldsA + (unsigned)((cur ^ 1) * TILEB * 2), Abase + (kt + 1) * 32, 32, 128, K, 3);
      tdm_load_2d(ldsB + (unsigned)((cur ^ 1) * TILEB * 2), Bbase + (kt + 1) * 32, 32, 128, K, 3);
    }
    const __bf16* Asc = &As[cur * TILEB];
    const __bf16* Bsc = &Bs[cur * TILEB];

    Frag af[4], bfr[2];
    for (int mi = 0; mi < 4; ++mi) {
      int row = wy * 64 + mi * 16 + n16;
      af[mi].h[0] = *(const bf16x8*)&Asc[row * LDSTRIDE + half * 8];
      af[mi].h[1] = *(const bf16x8*)&Asc[row * LDSTRIDE + 16 + half * 8];
    }
    for (int ni = 0; ni < 2; ++ni) {
      int col = wx * 32 + ni * 16 + n16;
      bfr[ni].h[0] = *(const bf16x8*)&Bsc[col * LDSTRIDE + half * 8];
      bfr[ni].h[1] = *(const bf16x8*)&Bsc[col * LDSTRIDE + 16 + half * 8];
    }
    for (int mi = 0; mi < 4; ++mi)
      for (int ni = 0; ni < 2; ++ni)
        acc[mi][ni] = __builtin_amdgcn_wmma_f32_16x16x32_bf16(
            false, af[mi].v, false, bfr[ni].v, (short)0, acc[mi][ni], false, false);
  }

  for (int mi = 0; mi < 4; ++mi)
    for (int ni = 0; ni < 2; ++ni)
      for (int r = 0; r < 8; ++r) {
        int row = m0 + wy * 64 + mi * 16 + half * 8 + r;
        int col = n0 + wx * 32 + ni * 16 + n16;
        C[(size_t)row * N + col] = acc[mi][ni][r];
      }
}

// ---------------- RoPE + split/scatter to [head][T][HS] bf16 ----------------
__global__ __launch_bounds__(256)
void rope_kernel(const float* __restrict__ qkv, const float* __restrict__ cosb,
                 const float* __restrict__ sinb, __bf16* __restrict__ Qb,
                 __bf16* __restrict__ Kb, __bf16* __restrict__ Vb) {
  int t = blockIdx.x;
  int tid = threadIdx.x;
  const float* row = qkv + (size_t)t * QKV_F;
  const float* cs = cosb + (size_t)t * HSZ;
  const float* sn = sinb + (size_t)t * HSZ;
  for (int i = tid; i < NHEAD * HSZ; i += 256) {
    int h = i >> 7, d = i & 127;
    float v = row[i];
    float rot = (d < 64) ? -row[i + 64] : row[i - 64];
    Qb[((size_t)h * T_SEQ + t) * HSZ + d] = f2bf(v * cs[d] + rot * sn[d]);
  }
  for (int i = tid; i < G_KV * HSZ; i += 256) {
    int g = i >> 7, d = i & 127;
    float v = row[NHEAD * HSZ + i];
    float rot = (d < 64) ? -row[NHEAD * HSZ + i + 64] : row[NHEAD * HSZ + i - 64];
    Kb[((size_t)g * T_SEQ + t) * HSZ + d] = f2bf(v * cs[d] + rot * sn[d]);
  }
  for (int i = tid; i < G_KV * HSZ; i += 256) {
    int g = i >> 7, d = i & 127;
    Vb[((size_t)g * T_SEQ + t) * HSZ + d] = f2bf(row[(NHEAD + G_KV) * HSZ + i]);
  }
}

// ---------------- Flash attention (causal, GQA) ----------------
// grid(NH, T/64), 128 threads = 4 waves; wave w owns 16 query rows.
// K tiles staged by TDM; V staged transposed by the waves (TDM can't transpose).
__global__ __launch_bounds__(128)
void attn_kernel(const __bf16* __restrict__ Qb, const __bf16* __restrict__ Kb,
                 const __bf16* __restrict__ Vb, __bf16* __restrict__ Yb) {
  __shared__ __align__(16) __bf16 Ks[32 * 136];          // [key][d], stride 272B
  __shared__ __align__(16) __bf16 Vt[128 * LDSTRIDE];    // [d][key], stride 80B
  __shared__ __align__(16) __bf16 Ps[4 * 16 * LDSTRIDE]; // per-wave P tile

  const float scale = 0.08838834764831845f; // 1/sqrt(128)
  int head = blockIdx.x, qblk = blockIdx.y;
  int tid = threadIdx.x;
  int wave = tid >> 5, lane = tid & 31, n16 = lane & 15, half = lane >> 4;
  int g = head >> 2; // Q_PER_KV = 4
  int q0 = qblk * 64 + wave * 16;

  // Q fragments (16 rows x 128) : 4 K-chunks of 32
  Frag aq[4];
  {
    const __bf16* qrow = Qb + ((size_t)head * T_SEQ + q0 + n16) * HSZ;
    for (int c = 0; c < 4; ++c) {
      aq[c].h[0] = *(const bf16x8*)(qrow + c * 32 + half * 8);
      aq[c].h[1] = *(const bf16x8*)(qrow + c * 32 + 16 + half * 8);
    }
  }

  float m[8], l[8];
  floatx8 yacc[8];
  for (int r = 0; r < 8; ++r) { m[r] = -3.0e38f; l[r] = 0.f; }
  for (int hh = 0; hh < 8; ++hh) yacc[hh] = (floatx8)0.f;

  int nkb_blk = 2 * qblk + 2;              // key blocks this workgroup stages
  int nkb_my  = (q0 + 16 + 31) >> 5;       // key blocks this wave computes
  __bf16* pw = &Ps[wave * 16 * LDSTRIDE];  // wave-private P region
  unsigned ldsK = (unsigned)(uintptr_t)&Ks[0];

  for (int kb = 0; kb < nkb_blk; ++kb) {
    __syncthreads(); // previous iteration's compute done
    if (wave == 0) { // TDM: 32 keys x 128 d, rows 256B -> pad_interval=5
      tdm_load_2d(ldsK, Kb + ((size_t)g * T_SEQ + kb * 32) * HSZ, 128, 32, HSZ, 5);
    }
    { // cooperative V tile load, transposed into Vt[d][key]
      int r = tid >> 2;
      int seg = (tid & 3) * 32;
      const __bf16* src = Vb + ((size_t)g * T_SEQ + kb * 32 + r) * HSZ + seg;
      for (int j = 0; j < 32; ++j) Vt[(seg + j) * LDSTRIDE + r] = src[j];
    }
    if (wave == 0) __builtin_amdgcn_s_wait_tensorcnt(0);
    __syncthreads();
    if (kb >= nkb_my) continue; // wave-uniform: EXEC stays all-ones

    // S = Q K^T for 32 keys (two 16-key subtiles)
    floatx8 s0 = (floatx8)0.f, s1 = (floatx8)0.f;
    for (int c = 0; c < 4; ++c) {
      Frag bk0, bk1;
      bk0.h[0] = *(const bf16x8*)&Ks[n16 * 136 + c * 32 + half * 8];
      bk0.h[1] = *(const bf16x8*)&Ks[n16 * 136 + c * 32 + 16 + half * 8];
      bk1.h[0] = *(const bf16x8*)&Ks[(16 + n16) * 136 + c * 32 + half * 8];
      bk1.h[1] = *(const bf16x8*)&Ks[(16 + n16) * 136 + c * 32 + 16 + half * 8];
      s0 = __builtin_amdgcn_wmma_f32_16x16x32_bf16(false, aq[c].v, false, bk0.v,
                                                   (short)0, s0, false, false);
      s1 = __builtin_amdgcn_wmma_f32_16x16x32_bf16(false, aq[c].v, false, bk1.v,
                                                   (short)0, s1, false, false);
    }

    // online softmax (rows live in VGPR index r, spread over 16-lane halves)
    int k0i = kb * 32 + n16, k1i = k0i + 16;
    for (int r = 0; r < 8; ++r) {
      int rr = q0 + half * 8 + r;
      float sr0 = s0[r] * scale, sr1 = s1[r] * scale;
      if (k0i > rr) sr0 = -3.0e38f;
      if (k1i > rr) sr1 = -3.0e38f;
      float mx = redmax16(fmaxf(sr0, sr1));
      mx = fmaxf(mx, m[r]);
      float a0 = __expf(sr0 - mx);
      float a1 = __expf(sr1 - mx);
      float sc = __expf(m[r] - mx);
      m[r] = mx;
      l[r] = l[r] * sc + redsum16(a0 + a1);
      for (int hh = 0; hh < 8; ++hh) yacc[hh][r] *= sc;
      pw[(half * 8 + r) * LDSTRIDE + n16]      = f2bf(a0);
      pw[(half * 8 + r) * LDSTRIDE + 16 + n16] = f2bf(a1);
    }

    // transpose P (D layout -> A layout) through wave-private LDS; same-wave
    // DS ops are in-order, compiler inserts s_wait_dscnt for the loads.
    Frag ap;
    ap.h[0] = *(const bf16x8*)&pw[n16 * LDSTRIDE + half * 8];
    ap.h[1] = *(const bf16x8*)&pw[n16 * LDSTRIDE + 16 + half * 8];

    // Y += P V  (8 HS tiles of 16)
    for (int hh = 0; hh < 8; ++hh) {
      Frag bv;
      bv.h[0] = *(const bf16x8*)&Vt[(hh * 16 + n16) * LDSTRIDE + half * 8];
      bv.h[1] = *(const bf16x8*)&Vt[(hh * 16 + n16) * LDSTRIDE + 16 + half * 8];
      yacc[hh] = __builtin_amdgcn_wmma_f32_16x16x32_bf16(false, ap.v, false, bv.v,
                                                         (short)0, yacc[hh], false, false);
    }
  }

  // epilogue: normalize, write Y[T, NH*HS] bf16
  for (int hh = 0; hh < 8; ++hh)
    for (int r = 0; r < 8; ++r) {
      int row = q0 + half * 8 + r;
      int col = head * HSZ + hh * 16 + n16;
      Yb[(size_t)row * (NHEAD * HSZ) + col] = f2bf(yacc[hh][r] / l[r]);
    }
}

extern "C" void kernel_launch(void* const* d_in, const int* in_sizes, int n_in,
                              void* d_out, int out_size, void* d_ws, size_t ws_size,
                              hipStream_t stream) {
  const float* x      = (const float*)d_in[0];
  const float* cosb   = (const float*)d_in[1];
  const float* sinb   = (const float*)d_in[2];
  const float* qkv_w  = (const float*)d_in[3];
  const float* proj_w = (const float*)d_in[4];
  float* out = (float*)d_out;

  char* ws = (char*)d_ws;
  __bf16* xb     = (__bf16*)ws; ws += (size_t)T_SEQ * C_DIM * 2;
  __bf16* wqkvb  = (__bf16*)ws; ws += (size_t)QKV_F * C_DIM * 2;
  __bf16* wprojb = (__bf16*)ws; ws += (size_t)C_DIM * C_DIM * 2;
  float*  qkvf   = (float*)ws;  ws += (size_t)T_SEQ * QKV_F * 4;
  __bf16* Qb     = (__bf16*)ws; ws += (size_t)NHEAD * T_SEQ * HSZ * 2;
  __bf16* Kb     = (__bf16*)ws; ws += (size_t)G_KV * T_SEQ * HSZ * 2;
  __bf16* Vb     = (__bf16*)ws; ws += (size_t)G_KV * T_SEQ * HSZ * 2;
  __bf16* Yb     = (__bf16*)ws; ws += (size_t)T_SEQ * C_DIM * 2;

  f32_to_bf16_kernel<<<2048, 256, 0, stream>>>(x, xb, T_SEQ * C_DIM);
  f32_to_bf16_kernel<<<3072, 256, 0, stream>>>(qkv_w, wqkvb, QKV_F * C_DIM);
  f32_to_bf16_kernel<<<2048, 256, 0, stream>>>(proj_w, wprojb, C_DIM * C_DIM);

  dim3 g1(QKV_F / 128, T_SEQ / 128);
  gemm_bf16_f32<<<g1, 256, 0, stream>>>(xb, wqkvb, qkvf, T_SEQ, QKV_F, C_DIM);

  rope_kernel<<<T_SEQ, 256, 0, stream>>>(qkvf, cosb, sinb, Qb, Kb, Vb);

  dim3 g2(NHEAD, T_SEQ / 64);
  attn_kernel<<<g2, 128, 0, stream>>>(Qb, Kb, Vb, Yb);

  dim3 g3(C_DIM / 128, T_SEQ / 128);
  gemm_bf16_f32<<<g3, 256, 0, stream>>>(Yb, wprojb, out, T_SEQ, C_DIM, C_DIM);
}
